// MolGNN2_23467701305420
// MI455X (gfx1250) — matrix operand order repaired and checked by reading, not compiled, not run on hardware
//
#include <hip/hip_runtime.h>
#include <hip/hip_bf16.h>

#define USE_ASYNC_B 1

// ---------------------------------------------------------------------------
// Types & helpers
// ---------------------------------------------------------------------------
typedef __attribute__((ext_vector_type(16))) __bf16 v16bf;
typedef __attribute__((ext_vector_type(8)))  float  v8f;

union FragBF { unsigned int w[8]; v16bf v; };

__device__ __forceinline__ unsigned short f2bf(float f) {
  unsigned int u = __float_as_uint(f);
  u += 0x7FFFu + ((u >> 16) & 1u);           // round-to-nearest-even
  return (unsigned short)(u >> 16);
}

__device__ __forceinline__ unsigned int pack2(float a, float b) {
  return (unsigned int)f2bf(a) | ((unsigned int)f2bf(b) << 16);
}

__device__ __forceinline__ v8f zero8() {
  v8f z = {0.f, 0.f, 0.f, 0.f, 0.f, 0.f, 0.f, 0.f};
  return z;
}

__device__ __forceinline__ v8f wmma_bf16(const FragBF& a, const FragBF& b, v8f c) {
  // D = A(16x32 bf16) * B(32x16 bf16) + C(16x16 f32)
  return __builtin_amdgcn_wmma_f32_16x16x32_bf16(false, a.v, false, b.v,
                                                 (short)0, c, false, false);
}

// ---------------------------------------------------------------------------
// Weight conversion kernels (fp32 -> bf16, padded to Kpad, N-major layout)
// Bt[n*Kpad + k] holds B[k][n]
// ---------------------------------------------------------------------------

// src is [batch][Nc][K] row-major ("W.T usage": B[k][n] = src[n][k])
__global__ void pad_bf16_kernel(const float* __restrict__ src, int batch, int Nc,
                                int K, int Kpad, unsigned short* __restrict__ dst) {
  long long i = (long long)blockIdx.x * blockDim.x + threadIdx.x;
  long long tot = (long long)batch * Nc * Kpad;
  if (i >= tot) return;
  int k = (int)(i % Kpad);
  long long bn = i / Kpad;           // b*Nc + n
  dst[i] = (k < K) ? f2bf(src[bn * K + k]) : (unsigned short)0;
}

// src is [batch][K][Nc] row-major ("direct usage": B[k][n] = src[k][n])
__global__ void tpad_bf16_kernel(const float* __restrict__ src, int batch, int Nc,
                                 int K, int Kpad, unsigned short* __restrict__ dst) {
  long long i = (long long)blockIdx.x * blockDim.x + threadIdx.x;
  long long tot = (long long)batch * Nc * Kpad;
  if (i >= tot) return;
  int k = (int)(i % Kpad);
  long long r = i / Kpad;
  int n = (int)(r % Nc);
  int b = (int)(r / Nc);
  dst[i] = (k < K) ? f2bf(src[((long long)b * K + k) * Nc + n]) : (unsigned short)0;
}

// ---------------------------------------------------------------------------
// Generic GEMM: Out[M,Nc] = relu?( A[M,K](f32) * B(bf16, Bt layout) + bias )
// Block: 256 threads (8 waves). Block tile 128(M) x 64(N). K chunked by 32.
// A staged via vectorized f32->bf16 (fast path) ; B staged via
// global_load_async_to_lds_b128 (fast path, raw bf16, ASYNCcnt-tracked).
// ---------------------------------------------------------------------------
__global__ __launch_bounds__(256) void gemm_bf16_kernel(
    const float* __restrict__ A, int M, int K, int Kpad,
    const unsigned short* __restrict__ Bt, int Nc,
    const float* __restrict__ bias, float* __restrict__ Out, int doRelu) {
  __shared__ unsigned short Alds[128][36];   // 72B rows: 8B-aligned at k%4==0
  __shared__ unsigned short Blds[64][40];    // 80B rows: 16B-aligned at k%8==0
  const int t = threadIdx.x;
  const int wave = t >> 5, lane = t & 31;
  const int m16 = lane & 15, half = lane >> 4;
  const int blockM = blockIdx.x * 128;
  const int blockN = blockIdx.y * 64;
  const bool fullM = (blockM + 128 <= M);
  const bool fullN = (blockN + 64 <= Nc);

  v8f acc[4];
#pragma unroll
  for (int nt = 0; nt < 4; ++nt) acc[nt] = zero8();

  for (int kc = 0; kc < Kpad; kc += 32) {
    // ---- Stage A tile 128x32 (fp32 -> bf16) ----
    if (fullM && (kc + 32 <= K)) {
#pragma unroll
      for (int i = 0; i < 4; ++i) {
        int idx = t + i * 256;               // 1024 float4 in tile
        int r = idx >> 3;                    // 8 float4 per row
        int k = (idx & 7) * 4;
        const float* gp = &A[(size_t)(blockM + r) * K + kc + k];
        float4 q = *(const float4*)gp;
        uint2 u;
        u.x = pack2(q.x, q.y);
        u.y = pack2(q.z, q.w);
        *(uint2*)&Alds[r][k] = u;
        if (i == 0 && (kc + 64 <= K))
          __builtin_prefetch(gp + 32, 0, 0); // next K-chunk
      }
    } else {
#pragma unroll
      for (int i = 0; i < 16; ++i) {
        int idx = t + i * 256;
        int r = idx >> 5, k = idx & 31;
        int gr = blockM + r, gk = kc + k;
        float v = (gr < M && gk < K) ? A[(size_t)gr * K + gk] : 0.f;
        Alds[r][k] = f2bf(v);
      }
    }
    // ---- Stage B tile 64(n) x 32(k), already bf16 in Bt (n-major) ----
#if USE_ASYNC_B
    if (fullN) {
      // 256 lanes x 16B = whole 4KB tile in one async instruction per lane
      int n = t >> 2, seg = t & 3;           // 4 x 16B segments per 64B row
      const unsigned short* gp = &Bt[(size_t)(blockN + n) * Kpad + kc + 8 * seg];
      unsigned int ldsoff = (unsigned int)(size_t)&Blds[n][8 * seg];
      asm volatile("global_load_async_to_lds_b128 %0, %1, off"
                   :: "v"(ldsoff), "v"(gp)
                   : "memory");
    } else
#endif
    {
#pragma unroll
      for (int i = 0; i < 4; ++i) {
        int idx = t + i * 256;               // 1024 dword pairs
        int n = idx >> 4, kp = idx & 15;
        unsigned int w = 0;
        int gn = blockN + n;
        if (gn < Nc)
          w = *(const unsigned int*)&Bt[(size_t)gn * Kpad + kc + 2 * kp];
        *(unsigned int*)&Blds[n][2 * kp] = w;
      }
    }
#if USE_ASYNC_B
    asm volatile("s_wait_asynccnt 0x0" ::: "memory");
#endif
    __syncthreads();

    // A fragment (contiguous k = 16*half + s, matching B)
    FragBF a;
    const int arow = wave * 16 + m16;
#pragma unroll
    for (int v = 0; v < 8; ++v) {
      int k0 = 2 * v + 16 * half;
      a.w[v] = *(const unsigned int*)&Alds[arow][k0];
    }
#pragma unroll
    for (int nt = 0; nt < 4; ++nt) {
      FragBF b;
#pragma unroll
      for (int v = 0; v < 8; ++v) {
        int k0 = 2 * v + 16 * half;
        b.w[v] = *(const unsigned int*)&Blds[nt * 16 + m16][k0];
      }
      acc[nt] = wmma_bf16(a, b, acc[nt]);
    }
    __syncthreads();
  }

  // Store C (16x16 f32 layout: lane = 16*half + n, VGPR v -> row v + 8*half)
#pragma unroll
  for (int nt = 0; nt < 4; ++nt) {
    int col = blockN + nt * 16 + m16;
    if (col >= Nc) continue;
    float bv = bias ? bias[col] : 0.f;
#pragma unroll
    for (int v = 0; v < 8; ++v) {
      int row = blockM + wave * 16 + v + 8 * half;
      if (row < M) {
        float x = acc[nt][v] + bv;
        if (doRelu) x = fmaxf(x, 0.f);
        Out[(size_t)row * Nc + col] = x;
      }
    }
  }
}

// ---------------------------------------------------------------------------
// Fused GRU layer: gi = agg@Wih^T + bih, gh = h@Whh^T + bhh (both via WMMA,
// gate pre-activations staged in LDS fp32), then GRU update of h in place.
// Block: 256 threads, 16 node rows. 68 = 34(gi)+34(gh) 16x16 N-tiles.
// ---------------------------------------------------------------------------
__global__ __launch_bounds__(256) void gru_layer_kernel(
    const float* __restrict__ agg, float* __restrict__ h,
    const unsigned short* __restrict__ Biht, const unsigned short* __restrict__ Bhht,
    const float* __restrict__ bih, const float* __restrict__ bhh, int N) {
  __shared__ unsigned short A1[16][196];   // agg rows (bf16), 392B rows (8B align)
  __shared__ unsigned short A2[16][196];   // h rows (bf16)
  __shared__ float G[16][544];             // gi
  __shared__ float Hs[16][544];            // gh
  const int t = threadIdx.x;
  const int wave = t >> 5, lane = t & 31;
  const int m16 = lane & 15, half = lane >> 4;
  const int rowbase = blockIdx.x * 16;

  // Stage agg & h (fp32 -> bf16)
  if (rowbase + 16 <= N) {
    for (int i = t; i < 16 * 45; i += 256) {   // 45 float4 per 180-wide row
      int r = i / 45, j = i - r * 45;
      int k = 4 * j;
      size_t off = (size_t)(rowbase + r) * 180 + k;
      float4 qa = *(const float4*)&agg[off];
      float4 qh = *(const float4*)&h[off];
      uint2 ua, uh;
      ua.x = pack2(qa.x, qa.y); ua.y = pack2(qa.z, qa.w);
      uh.x = pack2(qh.x, qh.y); uh.y = pack2(qh.z, qh.w);
      *(uint2*)&A1[r][k] = ua;
      *(uint2*)&A2[r][k] = uh;
    }
  } else {
    for (int i = t; i < 16 * 180; i += 256) {
      int r = i / 180, k = i - r * 180;
      int node = rowbase + r;
      float v1 = 0.f, v2 = 0.f;
      if (node < N) {
        v1 = agg[(size_t)node * 180 + k];
        v2 = h[(size_t)node * 180 + k];
      }
      A1[r][k] = f2bf(v1);
      A2[r][k] = f2bf(v2);
    }
  }
  // zero-pad K 180..191
  for (int i = t; i < 16 * 6; i += 256) {
    int r = i / 6, j = i - r * 6;
    *(unsigned int*)&A1[r][180 + 2 * j] = 0u;
    *(unsigned int*)&A2[r][180 + 2 * j] = 0u;
  }
  __syncthreads();

  for (int tt = wave; tt < 68; tt += 8) {
    const int isGh = (tt >= 34);
    const int ntile = isGh ? (tt - 34) : tt;
    const unsigned short(*Asrc)[196] = isGh ? A2 : A1;
    const unsigned short* Bsrc = isGh ? Bhht : Biht;
    const int ng = ntile * 16 + m16;       // 0..543 (>=540 is pad)

    v8f acc = zero8();
    for (int kc = 0; kc < 192; kc += 32) {
      FragBF a, b;
#pragma unroll
      for (int v = 0; v < 8; ++v) {
        int k0 = kc + 2 * v + 16 * half;
        a.w[v] = *(const unsigned int*)&Asrc[m16][k0];
      }
      if (ng < 540) {
#pragma unroll
        for (int v = 0; v < 8; ++v) {
          int k0 = kc + 2 * v + 16 * half;
          b.w[v] = *(const unsigned int*)&Bsrc[(size_t)ng * 192 + k0];
        }
      } else {
#pragma unroll
        for (int v = 0; v < 8; ++v) b.w[v] = 0u;
      }
      acc = wmma_bf16(a, b, acc);
    }
    float bv = (ng < 540) ? (isGh ? bhh[ng] : bih[ng]) : 0.f;
    float(*Gdst)[544] = isGh ? Hs : G;
#pragma unroll
    for (int v = 0; v < 8; ++v)
      Gdst[v + 8 * half][ng] = acc[v] + bv;
  }
  __syncthreads();

  // GRU elementwise update (fp32 state)
  for (int i = t; i < 16 * 180; i += 256) {
    int r = i / 180, c = i - r * 180;
    int node = rowbase + r;
    if (node < N) {
      float ir = G[r][c], iz = G[r][c + 180], in_ = G[r][c + 360];
      float hr = Hs[r][c], hz = Hs[r][c + 180], hn = Hs[r][c + 360];
      float rg = 1.f / (1.f + __expf(-(ir + hr)));
      float zg = 1.f / (1.f + __expf(-(iz + hz)));
      float nn = tanhf(in_ + rg * hn);
      float ho = h[(size_t)node * 180 + c];
      h[(size_t)node * 180 + c] = (1.f - zg) * nn + zg * ho;
    }
  }
}

// ---------------------------------------------------------------------------
// Misc element kernels
// ---------------------------------------------------------------------------
__global__ void init_h_kernel(const float* __restrict__ x, int N, float* __restrict__ h) {
  long long i = (long long)blockIdx.x * blockDim.x + threadIdx.x;
  if (i < (long long)N * 180) {
    int n = (int)(i / 180), c = (int)(i % 180);
    h[i] = (c < 64) ? x[(size_t)n * 64 + c] : 0.f;
  }
}

// one block per edge, 192 threads (180 active)
__global__ void scatter_add_kernel(const float* __restrict__ m,
                                   const int* __restrict__ ei, int E,
                                   float* __restrict__ agg) {
  int e = blockIdx.x;
  int c = threadIdx.x;
  if (e < E && c < 180) {
    int s = ei[e];        // src row
    int d = ei[E + e];    // dst row
    atomicAdd(&agg[(size_t)d * 180 + c], m[(size_t)s * 180 + c]);
  }
}

__global__ void pool_accum_kernel(const float* __restrict__ h,
                                  const int* __restrict__ batch, int N, int comp,
                                  float* __restrict__ pooled) {
  long long i = (long long)blockIdx.x * blockDim.x + threadIdx.x;
  if (i < (long long)N * 180) {
    int n = (int)(i / 180), c = (int)(i % 180);
    int g = batch[n];
    atomicAdd(&pooled[(size_t)g * 540 + comp * 180 + c], fmaxf(h[i], 0.f));
  }
}

__global__ void count_kernel(const int* __restrict__ batch, int N, float* __restrict__ cnt) {
  int n = blockIdx.x * blockDim.x + threadIdx.x;
  if (n < N) atomicAdd(&cnt[batch[n]], 1.f);
}

__global__ void bn_finalize_kernel(const float* __restrict__ pooled,
                                   const float* __restrict__ cnt,
                                   const float* __restrict__ gamma,
                                   const float* __restrict__ beta,
                                   const float* __restrict__ mean,
                                   const float* __restrict__ var,
                                   float* __restrict__ xbn) {
  int i = blockIdx.x * blockDim.x + threadIdx.x;
  if (i < 2048 * 540) {
    int g = i / 540, col = i % 540, comp = col / 180;
    float c = fmaxf(cnt[comp * 2048 + g], 1.f);
    float x = pooled[i] / c;
    x = (x - mean[col]) * rsqrtf(var[col] + 1e-5f) * gamma[col] + beta[col];
    xbn[i] = x;
  }
}

__global__ void fc3_kernel(const float* __restrict__ y, const float* __restrict__ W,
                           const float* __restrict__ b, float* __restrict__ out) {
  int i = blockIdx.x * blockDim.x + threadIdx.x;
  if (i < 2048 * 3) {
    int g = i / 3, o = i % 3;
    float s = b[o];
    for (int k = 0; k < 540; ++k)
      s += y[(size_t)g * 540 + k] * W[o * 540 + k];
    out[i] = s;
  }
}

// ---------------------------------------------------------------------------
// Host orchestration
// ---------------------------------------------------------------------------
extern "C" void kernel_launch(void* const* d_in, const int* in_sizes, int n_in,
                              void* d_out, int out_size, void* d_ws, size_t ws_size,
                              hipStream_t stream) {
  (void)n_in; (void)out_size; (void)ws_size;
  const int NG = 2048;

  // Inputs (setup_inputs() dict order)
  const float* xs[3]   = {(const float*)d_in[0], (const float*)d_in[3], (const float*)d_in[6]};
  const int*   eis[3]  = {(const int*)d_in[1],   (const int*)d_in[4],   (const int*)d_in[7]};
  const int*   bats[3] = {(const int*)d_in[2],   (const int*)d_in[5],   (const int*)d_in[8]};
  const float* conv_W  = (const float*)d_in[9];
  const float* gWih    = (const float*)d_in[10];
  const float* gWhh    = (const float*)d_in[11];
  const float* gbih    = (const float*)d_in[12];
  const float* gbhh    = (const float*)d_in[13];
  const float* bn_g    = (const float*)d_in[14];
  const float* bn_b    = (const float*)d_in[15];
  const float* bn_m    = (const float*)d_in[16];
  const float* bn_v    = (const float*)d_in[17];
  const float* fc1_W   = (const float*)d_in[18];
  const float* fc1_b   = (const float*)d_in[19];
  const float* fc2_W   = (const float*)d_in[20];
  const float* fc2_b   = (const float*)d_in[21];
  const float* fc25_W  = (const float*)d_in[22];
  const float* fc25_b  = (const float*)d_in[23];
  const float* fc3_W   = (const float*)d_in[24];
  const float* fc3_b   = (const float*)d_in[25];
  float* out = (float*)d_out;

  int Nc_[3], Ec_[3];
  int maxN = 0;
  for (int c = 0; c < 3; ++c) {
    Nc_[c] = in_sizes[3 * c] / 64;
    Ec_[c] = in_sizes[3 * c + 1] / 2;
    if (Nc_[c] > maxN) maxN = Nc_[c];
  }

  // ---- workspace carve ----
  char* p = (char*)d_ws;
  auto carve = [&](size_t bytes) -> void* {
    void* r = (void*)p;
    p += (bytes + 255) & ~(size_t)255;
    return r;
  };
  float* h      = (float*)carve((size_t)maxN * 180 * 4);
  float* m      = (float*)carve((size_t)maxN * 180 * 4);
  float* agg    = (float*)carve((size_t)maxN * 180 * 4);
  float* pooled = (float*)carve((size_t)NG * 540 * 4);
  float* cnt    = (float*)carve((size_t)3 * NG * 4);
  float* xbn    = (float*)carve((size_t)NG * 540 * 4);
  float* y1     = (float*)carve((size_t)NG * 1620 * 4);
  float* y2     = (float*)carve((size_t)NG * 1620 * 4);
  float* y25    = (float*)carve((size_t)NG * 540 * 4);
  unsigned short* Bconv = (unsigned short*)carve((size_t)18 * 180 * 192 * 2);
  unsigned short* Bih   = (unsigned short*)carve((size_t)3 * 540 * 192 * 2);
  unsigned short* Bhh   = (unsigned short*)carve((size_t)3 * 540 * 192 * 2);
  unsigned short* Bfc1  = (unsigned short*)carve((size_t)1620 * 544 * 2);
  unsigned short* Bfc2  = (unsigned short*)carve((size_t)1620 * 1632 * 2);
  unsigned short* Bfc25 = (unsigned short*)carve((size_t)540 * 1632 * 2);

  auto blocks = [](long long tot, int bs) { return (unsigned)((tot + bs - 1) / bs); };

  // ---- weight conversion (fp32 -> padded bf16, N-major) ----
  tpad_bf16_kernel<<<blocks((long long)18 * 180 * 192, 256), 256, 0, stream>>>(
      conv_W, 18, 180, 180, 192, Bconv);
  pad_bf16_kernel<<<blocks((long long)3 * 540 * 192, 256), 256, 0, stream>>>(
      gWih, 3, 540, 180, 192, Bih);
  pad_bf16_kernel<<<blocks((long long)3 * 540 * 192, 256), 256, 0, stream>>>(
      gWhh, 3, 540, 180, 192, Bhh);
  pad_bf16_kernel<<<blocks((long long)1620 * 544, 256), 256, 0, stream>>>(
      fc1_W, 1, 1620, 540, 544, Bfc1);
  pad_bf16_kernel<<<blocks((long long)1620 * 1632, 256), 256, 0, stream>>>(
      fc2_W, 1, 1620, 1620, 1632, Bfc2);
  pad_bf16_kernel<<<blocks((long long)540 * 1632, 256), 256, 0, stream>>>(
      fc25_W, 1, 540, 1620, 1632, Bfc25);

  // ---- zero pooled sums / counts ----
  hipMemsetAsync(pooled, 0, (size_t)NG * 540 * 4, stream);
  hipMemsetAsync(cnt, 0, (size_t)3 * NG * 4, stream);

  // ---- per graph-component GatedGraphConv ----
  for (int comp = 0; comp < 3; ++comp) {
    const int N = Nc_[comp], E = Ec_[comp];
    init_h_kernel<<<blocks((long long)N * 180, 256), 256, 0, stream>>>(xs[comp], N, h);
    for (int l = 0; l < 6; ++l) {
      // m = h @ W[l]
      gemm_bf16_kernel<<<dim3(blocks(N, 128), 3), 256, 0, stream>>>(
          h, N, 180, 192, Bconv + (size_t)(comp * 6 + l) * 180 * 192, 180,
          nullptr, m, 0);
      // agg = scatter_add(m[src] -> dst)
      hipMemsetAsync(agg, 0, (size_t)N * 180 * 4, stream);
      scatter_add_kernel<<<E, 192, 0, stream>>>(m, eis[comp], E, agg);
      // fused GRU (gi/gh WMMA + gates), updates h in place
      gru_layer_kernel<<<blocks(N, 16), 256, 0, stream>>>(
          agg, h, Bih + (size_t)comp * 540 * 192, Bhh + (size_t)comp * 540 * 192,
          gbih + comp * 540, gbhh + comp * 540, N);
    }
    // relu + segment-sum pooling + counts
    pool_accum_kernel<<<blocks((long long)N * 180, 256), 256, 0, stream>>>(
        h, bats[comp], N, comp, pooled);
    count_kernel<<<blocks(N, 256), 256, 0, stream>>>(bats[comp], N, cnt + comp * NG);
  }

  // ---- mean + BatchNorm ----
  bn_finalize_kernel<<<blocks((long long)NG * 540, 256), 256, 0, stream>>>(
      pooled, cnt, bn_g, bn_b, bn_m, bn_v, xbn);

  // ---- MLP head (WMMA GEMMs) ----
  gemm_bf16_kernel<<<dim3(blocks(NG, 128), blocks(1620, 64)), 256, 0, stream>>>(
      xbn, NG, 540, 544, Bfc1, 1620, fc1_b, y1, 1);
  gemm_bf16_kernel<<<dim3(blocks(NG, 128), blocks(1620, 64)), 256, 0, stream>>>(
      y1, NG, 1620, 1632, Bfc2, 1620, fc2_b, y2, 1);
  gemm_bf16_kernel<<<dim3(blocks(NG, 128), blocks(540, 64)), 256, 0, stream>>>(
      y2, NG, 1620, 1632, Bfc25, 540, fc25_b, y25, 1);
  fc3_kernel<<<blocks((long long)NG * 3, 256), 256, 0, stream>>>(y25, fc3_W, fc3_b, out);
}
